// P2BVoteNetRPN2_39127152067002
// MI455X (gfx1250) — compile-verified, matrix operand order checked by reading, not run on hardware
//
#include <hip/hip_runtime.h>
#include <hip/hip_bf16.h>
#include <math.h>

// ---------------------------------------------------------------------------
// P2B VoteNet RPN forward for MI455X (gfx1250, wave32, WMMA).
// f16 activations in [B][N][Cpad] layout (K-contiguous), f16 packed weights,
// guard-free b128 LDS staging, double-buffered tiles, f32 WMMA accumulation.
// Register-blocked waves: 2 A-frags x 4 B-frags = 8 WMMAs per K-step.
// ---------------------------------------------------------------------------

typedef __attribute__((ext_vector_type(16))) _Float16 v16h;
typedef __attribute__((ext_vector_type(8)))  _Float16 v8h;
typedef __attribute__((ext_vector_type(8)))  float    v8f;

#define Bt   32
#define Npt  2048
#define Ff   256
#define VCc  256
#define NPp  64
#define NSs  16
#define RAD2 (0.3f * 0.3f)

#define BM 64
#define BK 32
#define XFSTR 288   // xyz_feature / "new" padded channel stride
#define VSTR  320   // vote buffer padded channel stride (259 -> 320 = 5*BM)

// ===========================================================================
// WMMA GEMM (f16 in / f16 out, f32 accumulate):
//   Yh[b,n,m] = epilogue( sum_k Wh[m,k] * Xh[b,n,k] )
//   mode 0: relu(acc*scale[m]+bias[m]);  mode 1: acc + bias[m] + resid[b,n,m]
// Block tile: BM=64(M) x BN_T(N) x 32(K), 8 waves = WMG x (8/WMG),
// per-wave tile: (MF*16) x (NF*16); double-buffered LDS.
// ===========================================================================
template<int BN_T, int WMG, int MF, int NF>
__global__ __launch_bounds__(256)
void wmma_gemm_t(const _Float16* __restrict__ Wh,   // Opad x Cpad (f16, padded)
                 const _Float16* __restrict__ Xh,   // B x Ncols x Cpad (f16)
                 _Float16* __restrict__ Yh,         // B x Ncols x Opad (f16)
                 const float* __restrict__ scale,   // O (mode 0)
                 const float* __restrict__ bias,    // O
                 const _Float16* __restrict__ resid,// B x Ncols x Rstride (mode 1)
                 int O, int Opad, int Cpad, int Ncols, int Rstride, int mode)
{
    __shared__ __align__(16) _Float16 Ws[2][BM][BK + 8];
    __shared__ __align__(16) _Float16 Xs[2][BN_T][BK + 8];

    const int b   = blockIdx.z;
    const int m0  = blockIdx.y * BM;
    const int n0  = blockIdx.x * BN_T;
    const int tid = threadIdx.x;
    const int wave = tid >> 5, lane = tid & 31;
    const int wm = wave % WMG;             // wave M group (MF*16 rows)
    const int wn = wave / WMG;             // wave N group (NF*16 cols)
    const int lrow  = lane & 15;
    const int khalf = (lane >> 4) << 3;    // K sub-offset 0/8 (16-bit layout)
    const int mrow8 = (lane >> 4) << 3;    // C/D row offset 0/8

    const _Float16* Xb = Xh + (size_t)b * Ncols * Cpad;

    // staging decomposition: one v8h (16B) per thread per chunk
    const int wr = tid >> 2, wq = tid & 3;           // tiles: rows x 4 chunks
    const int ksteps = Cpad / BK;

    auto stage = [&](int bb, int ks) {
        const int k0 = ks * BK;
        *(v8h*)&Ws[bb][wr][wq * 8] =
            *(const v8h*)(Wh + (size_t)(m0 + wr) * Cpad + k0 + wq * 8);
        #pragma unroll
        for (int it = 0; it < BN_T / 64; ++it) {
            int idx = it * 256 + tid;
            int xr = idx >> 2, xq = idx & 3;
            *(v8h*)&Xs[bb][xr][xq * 8] =
                *(const v8h*)(Xb + (size_t)(n0 + xr) * Cpad + k0 + xq * 8);
        }
    };

    v8f acc[MF][NF] = {};

    stage(0, 0);
    __syncthreads();
    for (int ks = 0; ks < ksteps; ++ks) {
        const int cur = ks & 1;
        if (ks + 1 < ksteps) stage(cur ^ 1, ks + 1);

        // A fragments (weights), per ISA 16-bit A layout
        v16h a[MF];
        #pragma unroll
        for (int mf = 0; mf < MF; ++mf) {
            const _Float16* wp = &Ws[cur][wm * (MF * 16) + mf * 16 + lrow][0];
            v8h alo = *(const v8h*)(wp + khalf);
            v8h ahi = *(const v8h*)(wp + 16 + khalf);
            a[mf] = __builtin_shufflevector(alo, ahi, 0, 1, 2, 3, 4, 5, 6, 7,
                                                      8, 9, 10, 11, 12, 13, 14, 15);
        }
        // B fragments (activations)
        v16h bfr[NF];
        #pragma unroll
        for (int nf = 0; nf < NF; ++nf) {
            const _Float16* xp = &Xs[cur][wn * (NF * 16) + nf * 16 + lrow][0];
            v8h blo = *(const v8h*)(xp + khalf);
            v8h bhi = *(const v8h*)(xp + 16 + khalf);
            bfr[nf] = __builtin_shufflevector(blo, bhi, 0, 1, 2, 3, 4, 5, 6, 7,
                                                        8, 9, 10, 11, 12, 13, 14, 15);
        }
        // burst of MF*NF WMMAs (independent accumulators)
        #pragma unroll
        for (int mf = 0; mf < MF; ++mf)
            #pragma unroll
            for (int nf = 0; nf < NF; ++nf)
                acc[mf][nf] = __builtin_amdgcn_wmma_f32_16x16x32_f16(
                    false, a[mf], false, bfr[nf], (short)0, acc[mf][nf], false, false);
        __syncthreads();
    }

    // ---- epilogue: lane owns 8 consecutive m at fixed n -> one b128 store ----
    _Float16* Yb = Yh + (size_t)b * Ncols * Opad;
    #pragma unroll
    for (int mf = 0; mf < MF; ++mf) {
        const int m_base = m0 + wm * (MF * 16) + mf * 16 + mrow8;
        float sc[8], bs[8];
        #pragma unroll
        for (int r = 0; r < 8; ++r) {
            int m = m_base + r;
            sc[r] = (mode == 0 && m < O) ? scale[m] : 0.0f;
            bs[r] = (m < O) ? bias[m] : 0.0f;
        }
        #pragma unroll
        for (int nf = 0; nf < NF; ++nf) {
            int n = n0 + wn * (NF * 16) + nf * 16 + lrow;
            v8h o;
            #pragma unroll
            for (int r = 0; r < 8; ++r) {
                int m = m_base + r;
                float v = 0.0f;
                if (m < O) {
                    if (mode == 0) {
                        v = acc[mf][nf][r] * sc[r] + bs[r];
                        v = v > 0.0f ? v : 0.0f;
                    } else {
                        v = acc[mf][nf][r] + bs[r] +
                            (float)resid[((size_t)b * Ncols + n) * Rstride + m];
                    }
                }
                o[r] = (_Float16)v;
            }
            *(v8h*)(Yb + (size_t)n * Opad + m_base) = o;
        }
    }
}

// ===========================================================================
// Packing / glue kernels
// ===========================================================================

// Pack f32 weight (O x C) -> f16 (Opad x Cpad), zero padded.
__global__ __launch_bounds__(256)
void pack_weight_kernel(const float* __restrict__ W, _Float16* __restrict__ out,
                        int O, int C, int Opad, int Cpad)
{
    int t = blockIdx.x * 256 + threadIdx.x;
    if (t >= Opad * Cpad) return;
    int c = t % Cpad, o = t / Cpad;
    out[t] = (o < O && c < C) ? (_Float16)W[(size_t)o * C + c] : (_Float16)0.0f;
}

// featT[b,n,c] = (f16) feature[b,c,n]   (C=256)
__global__ __launch_bounds__(256)
void pack_featT_kernel(const float* __restrict__ feat, _Float16* __restrict__ out)
{
    size_t t = (size_t)blockIdx.x * 256 + threadIdx.x;
    if (t >= (size_t)Bt * Npt * Ff) return;
    int c = (int)(t % Ff);
    int n = (int)((t / Ff) % Npt);
    int b = (int)(t / ((size_t)Ff * Npt));
    out[t] = (_Float16)feat[((size_t)b * Ff + c) * Npt + n];
}

// xyzfT[b,n,c] (stride XFSTR): c<3 -> xyz[b,n,c]; c<259 -> feature[b,c-3,n]; else 0
__global__ __launch_bounds__(256)
void pack_xyzfT_kernel(const float* __restrict__ xyz, const float* __restrict__ feat,
                       _Float16* __restrict__ out)
{
    size_t t = (size_t)blockIdx.x * 256 + threadIdx.x;
    if (t >= (size_t)Bt * Npt * XFSTR) return;
    int c = (int)(t % XFSTR);
    int n = (int)((t / XFSTR) % Npt);
    int b = (int)(t / ((size_t)XFSTR * Npt));
    float v = 0.0f;
    if (c < 3)        v = xyz[((size_t)b * Npt + n) * 3 + c];
    else if (c < 259) v = feat[((size_t)b * Ff + (c - 3)) * Npt + n];
    out[t] = (_Float16)v;
}

// out[b,j,o] = sum_c W[o,c] * Xh[b,j,c] + bb[o]   (tiny-O heads, f16 X)
__global__ __launch_bounds__(256)
void head_kernel(const float* __restrict__ Wt, const float* __restrict__ bb,
                 const _Float16* __restrict__ Xh, float* __restrict__ out,
                 int Ohead, int C, int Ncols)
{
    int t = blockIdx.x * 256 + threadIdx.x;
    if (t >= Bt * Ncols * Ohead) return;
    int o = t % Ohead;
    int j = (t / Ohead) % Ncols;
    int b = t / (Ohead * Ncols);
    const _Float16* xc = Xh + ((size_t)b * Ncols + j) * C;
    const float* wr = Wt + (size_t)o * C;
    float s = bb[o];
    #pragma unroll 8
    for (int c = 0; c < C; ++c) s += wr[c] * (float)xc[c];
    out[t] = s;
}

// vote_xyz[b,n,k] = (f32) vote[b,n,k]  (k<3, vote stride VSTR)
__global__ __launch_bounds__(256)
void pack_vxyz_kernel(const _Float16* __restrict__ vote, float* __restrict__ vx)
{
    int t = blockIdx.x * 256 + threadIdx.x;
    if (t >= Bt * Npt * 3) return;
    int k = t % 3;
    int n = (t / 3) % Npt;
    int b = t / (3 * Npt);
    vx[t] = (float)vote[((size_t)b * Npt + n) * VSTR + k];
}

// Farthest-point sampling: one block per batch, dists in registers (8/thread).
__global__ __launch_bounds__(256)
void fps_kernel(const float* __restrict__ vxyz, int* __restrict__ fps_idx)
{
    const int b = blockIdx.x;
    const int tid = threadIdx.x;
    const float* P = vxyz + (size_t)b * Npt * 3;
    float d[8];
    #pragma unroll
    for (int j = 0; j < 8; ++j) d[j] = 1e10f;
    __shared__ float sv[256];
    __shared__ int   si[256];
    __shared__ float spt[3];
    int last = 0;
    if (tid == 0) fps_idx[b * NPp] = 0;
    for (int step = 1; step < NPp; ++step) {
        if (tid < 3) spt[tid] = P[last * 3 + tid];
        __syncthreads();
        float px = spt[0], py = spt[1], pz = spt[2];
        float bestv = -1.0f;
        int besti = 0x7fffffff;
        #pragma unroll
        for (int j = 0; j < 8; ++j) {
            int n = tid + j * 256;
            float dx = P[n * 3 + 0] - px;
            float dy = P[n * 3 + 1] - py;
            float dz = P[n * 3 + 2] - pz;
            float dd = dx * dx + dy * dy + dz * dz;
            d[j] = fminf(d[j], dd);
            if (d[j] > bestv || (d[j] == bestv && n < besti)) { bestv = d[j]; besti = n; }
        }
        sv[tid] = bestv; si[tid] = besti;
        __syncthreads();
        for (int off = 128; off > 0; off >>= 1) {
            if (tid < off) {
                float v2 = sv[tid + off]; int i2 = si[tid + off];
                if (v2 > sv[tid] || (v2 == sv[tid] && i2 < si[tid])) {
                    sv[tid] = v2; si[tid] = i2;
                }
            }
            __syncthreads();
        }
        last = si[0];
        if (tid == 0) fps_idx[b * NPp + step] = last;
        __syncthreads();
    }
}

// centers gather + int64 fps idx output
__global__ __launch_bounds__(256)
void centers_kernel(const float* __restrict__ vxyz, const int* __restrict__ fi,
                    float* __restrict__ centers, long long* __restrict__ i64)
{
    int t = blockIdx.x * 256 + threadIdx.x;
    if (t >= Bt * NPp) return;
    int b = t / NPp;
    int id = fi[t];
    i64[t] = (long long)id;
    #pragma unroll
    for (int k = 0; k < 3; ++k)
        centers[t * 3 + k] = vxyz[((size_t)b * Npt + id) * 3 + k];
}

// Ball query: one thread per (b, center); first NS hits in index order.
__global__ __launch_bounds__(256)
void ball_query_kernel(const float* __restrict__ centers, const float* __restrict__ vxyz,
                       int* __restrict__ idx)
{
    int t = blockIdx.x * 256 + threadIdx.x;
    if (t >= Bt * NPp) return;
    int b = t / NPp;
    float cx = centers[t * 3 + 0], cy = centers[t * 3 + 1], cz = centers[t * 3 + 2];
    const float* P = vxyz + (size_t)b * Npt * 3;
    int* out = idx + (size_t)t * NSs;
    int cnt = 0, first = 0;
    for (int n = 0; n < Npt && cnt < NSs; ++n) {
        float dx = P[n * 3 + 0] - cx;
        float dy = P[n * 3 + 1] - cy;
        float dz = P[n * 3 + 2] - cz;
        if (dx * dx + dy * dy + dz * dz < RAD2) {
            if (cnt == 0) first = n;
            out[cnt++] = n;
        }
    }
    if (cnt == 0) { for (int s = 0; s < NSs; ++s) out[s] = 0; }
    else          { for (int s = cnt; s < NSs; ++s) out[s] = first; }
}

// Grouped tensor newT[b, j*16+s, c] (stride XFSTR, 260 real ch): block per (b,j).
// c<3: vxyz[id]-center; c==3: sigmoid(cla[id]); c<260: vote[b,id,c-1]; else 0.
__global__ __launch_bounds__(256)
void group_kernel(const float* __restrict__ vxyz, const float* __restrict__ centers,
                  const _Float16* __restrict__ vote, const float* __restrict__ cla,
                  const int* __restrict__ idx, _Float16* __restrict__ newt)
{
    int bj = blockIdx.x;
    int b = bj / NPp, j = bj % NPp;
    __shared__ int   sid[NSs];
    __shared__ float sc[3];
    if (threadIdx.x < NSs) sid[threadIdx.x] = idx[(size_t)bj * NSs + threadIdx.x];
    if (threadIdx.x < 3)   sc[threadIdx.x] = centers[bj * 3 + threadIdx.x];
    __syncthreads();
    for (int e = threadIdx.x; e < NSs * XFSTR; e += 256) {
        int c = e % XFSTR, s = e / XFSTR;
        int id = sid[s];
        float v = 0.0f;
        if (c < 3) {
            v = vxyz[((size_t)b * Npt + id) * 3 + c] - sc[c];
        } else if (c == 3) {
            float x = cla[(size_t)b * Npt + id];
            v = 1.0f / (1.0f + expf(-x));
        } else if (c < 260) {
            v = (float)vote[((size_t)b * Npt + id) * VSTR + (c - 1)];
        }
        newt[((size_t)b * (NPp * NSs) + j * NSs + s) * XFSTR + c] = (_Float16)v;
    }
}

// prop_feat[b,j,c] = max_s h[b, j*16+s, c]   (h stride 256)
__global__ __launch_bounds__(256)
void maxpool_kernel(const _Float16* __restrict__ h, _Float16* __restrict__ pf)
{
    int t = blockIdx.x * 256 + threadIdx.x;
    if (t >= Bt * NPp * VCc) return;
    int c = t % VCc;
    int j = (t / VCc) % NPp;
    int b = t / (VCc * NPp);
    const _Float16* hp = h + ((size_t)b * (NPp * NSs) + j * NSs) * VCc + c;
    float m = (float)hp[0];
    #pragma unroll
    for (int s = 1; s < NSs; ++s) m = fmaxf(m, (float)hp[(size_t)s * VCc]);
    pf[t] = (_Float16)m;
}

// ===========================================================================
// Host-side orchestration
// ===========================================================================
static inline int nb(long long total) { return (int)((total + 255) / 256); }

extern "C" void kernel_launch(void* const* d_in, const int* in_sizes, int n_in,
                              void* d_out, int out_size, void* d_ws, size_t ws_size,
                              hipStream_t stream)
{
    (void)in_sizes; (void)n_in; (void)out_size; (void)ws_size;
    const float* xyz      = (const float*)d_in[0];   // (B, N, 3)
    const float* feature  = (const float*)d_in[1];   // (B, 256, N)
    const float* cla_w1   = (const float*)d_in[2];
    const float* cla_s1   = (const float*)d_in[3];
    const float* cla_b1   = (const float*)d_in[4];
    const float* cla_w2   = (const float*)d_in[5];
    const float* cla_s2   = (const float*)d_in[6];
    const float* cla_b2   = (const float*)d_in[7];
    const float* cla_w3   = (const float*)d_in[8];
    const float* cla_bb3  = (const float*)d_in[9];
    const float* vote_w1  = (const float*)d_in[10];
    const float* vote_s1  = (const float*)d_in[11];
    const float* vote_b1  = (const float*)d_in[12];
    const float* vote_w2  = (const float*)d_in[13];
    const float* vote_s2  = (const float*)d_in[14];
    const float* vote_b2  = (const float*)d_in[15];
    const float* vote_w3  = (const float*)d_in[16];
    const float* vote_bb3 = (const float*)d_in[17];
    const float* sa_w1    = (const float*)d_in[18];
    const float* sa_s1    = (const float*)d_in[19];
    const float* sa_b1    = (const float*)d_in[20];
    const float* sa_w2    = (const float*)d_in[21];
    const float* sa_s2    = (const float*)d_in[22];
    const float* sa_b2    = (const float*)d_in[23];
    const float* sa_w3    = (const float*)d_in[24];
    const float* sa_s3    = (const float*)d_in[25];
    const float* sa_b3    = (const float*)d_in[26];
    const float* prop_w1  = (const float*)d_in[27];
    const float* prop_s1  = (const float*)d_in[28];
    const float* prop_b1  = (const float*)d_in[29];
    const float* prop_w2  = (const float*)d_in[30];
    const float* prop_s2  = (const float*)d_in[31];
    const float* prop_b2  = (const float*)d_in[32];
    const float* prop_w3  = (const float*)d_in[33];
    const float* prop_bb3 = (const float*)d_in[34];
    const float* score_w1  = (const float*)d_in[35];
    const float* score_s1  = (const float*)d_in[36];
    const float* score_b1  = (const float*)d_in[37];
    const float* score_w2  = (const float*)d_in[38];
    const float* score_s2  = (const float*)d_in[39];
    const float* score_b2  = (const float*)d_in[40];
    const float* score_w3  = (const float*)d_in[41];
    const float* score_bb3 = (const float*)d_in[42];

    // d_out regions (return order)
    float* out_est     = (float*)d_out;                       // 32*64*8
    float* out_cla     = out_est + 16384;                     // 32*2048
    float* out_vxyz    = out_cla + 65536;                     // 32*2048*3
    float* out_centers = out_vxyz + 196608;                   // 32*64*3
    float* out_score   = out_centers + 6144;                  // 32*64
    long long* out_i64 = (long long*)(out_score + 2048);      // 32*64 int64

    // Workspace layout (f16 activations)
    char* ws = (char*)d_ws;
    const size_t SZ_W     = (size_t)192 * 1024;                       // weight slot
    const size_t SZ_FEATT = (size_t)Bt * Npt * Ff * 2;                // 33.5 MB
    const size_t SZ_XYZF  = (size_t)Bt * Npt * XFSTR * 2;             // 37.7 MB
    const size_t SZ_VOTE  = (size_t)Bt * Npt * VSTR * 2;              // 42 MB

    _Float16* wp_slot[12];
    for (int i = 0; i < 12; ++i) wp_slot[i] = (_Float16*)(ws + i * SZ_W);
    char* p = ws + 12 * SZ_W;
    _Float16* featT = (_Float16*)p;              p += SZ_FEATT;  // later: newT
    _Float16* xyzfT = (_Float16*)p;              p += SZ_XYZF;
    _Float16* bufA  = (_Float16*)p;              p += SZ_FEATT;
    _Float16* bufB  = (_Float16*)p;              p += SZ_FEATT;
    _Float16* vote  = (_Float16*)p;              p += SZ_VOTE;   // later: h3
    _Float16* pf    = (_Float16*)p;              p += (size_t)Bt * NPp * VCc * 2;
    _Float16* ebuf1 = (_Float16*)p;              p += (size_t)Bt * NPp * VCc * 2;
    _Float16* ebuf2 = (_Float16*)p;              p += (size_t)Bt * NPp * VCc * 2;
    int* fps_i32    = (int*)p;                   p += (size_t)Bt * NPp * 4;
    int* ball_idx   = (int*)p;

    const dim3 blk(256);
    const dim3 gN2048(Npt / 256, 4, Bt);         // 256-out layers, N=2048
    const dim3 gVote3(Npt / 256, VSTR / BM, Bt); // 320-out (259 real)
    const dim3 gN1024((NPp * NSs) / 256, 4, Bt); // SA layers, N=1024
    const dim3 gN64(1, 4, Bt);                   // heads, N=64 (BN=64)

    // ---- pack weights to f16 padded ----
    struct { const float* w; int O, C, Opad, Cpad; } wl[12] = {
        {cla_w1, 256, 256, 256, 256},  {cla_w2, 256, 256, 256, 256},
        {vote_w1, 256, 259, 256, XFSTR}, {vote_w2, 256, 256, 256, 256},
        {vote_w3, 259, 256, VSTR, 256},
        {sa_w1, 256, 260, 256, XFSTR}, {sa_w2, 256, 256, 256, 256},
        {sa_w3, 256, 256, 256, 256},
        {prop_w1, 256, 256, 256, 256}, {prop_w2, 256, 256, 256, 256},
        {score_w1, 256, 256, 256, 256}, {score_w2, 256, 256, 256, 256},
    };
    for (int i = 0; i < 12; ++i)
        pack_weight_kernel<<<nb((long long)wl[i].Opad * wl[i].Cpad), blk, 0, stream>>>(
            wl[i].w, wp_slot[i], wl[i].O, wl[i].C, wl[i].Opad, wl[i].Cpad);

    // ---- pack activations ----
    pack_featT_kernel<<<nb((long long)Bt * Npt * Ff), blk, 0, stream>>>(feature, featT);
    pack_xyzfT_kernel<<<nb((long long)Bt * Npt * XFSTR), blk, 0, stream>>>(xyz, feature, xyzfT);

    // ---- cla branch ----
    wmma_gemm_t<256, 2, 2, 4><<<gN2048, blk, 0, stream>>>(wp_slot[0], featT, bufA,
        cla_s1, cla_b1, nullptr, 256, 256, 256, Npt, 0, 0);
    wmma_gemm_t<256, 2, 2, 4><<<gN2048, blk, 0, stream>>>(wp_slot[1], bufA, bufB,
        cla_s2, cla_b2, nullptr, 256, 256, 256, Npt, 0, 0);
    head_kernel<<<nb((long long)Bt * Npt), blk, 0, stream>>>(cla_w3, cla_bb3,
        bufB, out_cla, 1, 256, Npt);

    // ---- vote branch ----
    wmma_gemm_t<256, 2, 2, 4><<<gN2048, blk, 0, stream>>>(wp_slot[2], xyzfT, bufA,
        vote_s1, vote_b1, nullptr, 256, 256, XFSTR, Npt, 0, 0);
    wmma_gemm_t<256, 2, 2, 4><<<gN2048, blk, 0, stream>>>(wp_slot[3], bufA, bufB,
        vote_s2, vote_b2, nullptr, 256, 256, 256, Npt, 0, 0);
    wmma_gemm_t<256, 2, 2, 4><<<gVote3, blk, 0, stream>>>(wp_slot[4], bufB, vote,
        nullptr, vote_bb3, xyzfT, 259, VSTR, 256, Npt, XFSTR, 1);  // + residual
    pack_vxyz_kernel<<<nb((long long)Bt * Npt * 3), blk, 0, stream>>>(vote, out_vxyz);

    // ---- FPS / ball query / grouping ----
    fps_kernel<<<Bt, blk, 0, stream>>>(out_vxyz, fps_i32);
    centers_kernel<<<nb((long long)Bt * NPp), blk, 0, stream>>>(out_vxyz, fps_i32,
        out_centers, out_i64);
    ball_query_kernel<<<nb((long long)Bt * NPp), blk, 0, stream>>>(out_centers,
        out_vxyz, ball_idx);
    _Float16* newT = featT;   // reuse (B,1024,XFSTR)
    group_kernel<<<Bt * NPp, blk, 0, stream>>>(out_vxyz, out_centers, vote,
        out_cla, ball_idx, newT);

    // ---- SA MLPs + maxpool ----
    wmma_gemm_t<256, 2, 2, 4><<<gN1024, blk, 0, stream>>>(wp_slot[5], newT, bufA,
        sa_s1, sa_b1, nullptr, 256, 256, XFSTR, NPp * NSs, 0, 0);
    wmma_gemm_t<256, 2, 2, 4><<<gN1024, blk, 0, stream>>>(wp_slot[6], bufA, bufB,
        sa_s2, sa_b2, nullptr, 256, 256, 256, NPp * NSs, 0, 0);
    _Float16* h3 = vote;      // reuse vote region (stride 256 now)
    wmma_gemm_t<256, 2, 2, 4><<<gN1024, blk, 0, stream>>>(wp_slot[7], bufB, h3,
        sa_s3, sa_b3, nullptr, 256, 256, 256, NPp * NSs, 0, 0);
    maxpool_kernel<<<nb((long long)Bt * NPp * VCc), blk, 0, stream>>>(h3, pf);

    // ---- proposal head ----
    wmma_gemm_t<64, 4, 1, 2><<<gN64, blk, 0, stream>>>(wp_slot[8], pf, ebuf1,
        prop_s1, prop_b1, nullptr, 256, 256, 256, NPp, 0, 0);
    wmma_gemm_t<64, 4, 1, 2><<<gN64, blk, 0, stream>>>(wp_slot[9], ebuf1, ebuf2,
        prop_s2, prop_b2, nullptr, 256, 256, 256, NPp, 0, 0);
    head_kernel<<<nb((long long)Bt * NPp * 8), blk, 0, stream>>>(prop_w3, prop_bb3,
        ebuf2, out_est, 8, 256, NPp);   // first 8 of 9 rows

    // ---- score head ----
    wmma_gemm_t<64, 4, 1, 2><<<gN64, blk, 0, stream>>>(wp_slot[10], pf, ebuf1,
        score_s1, score_b1, nullptr, 256, 256, 256, NPp, 0, 0);
    wmma_gemm_t<64, 4, 1, 2><<<gN64, blk, 0, stream>>>(wp_slot[11], ebuf1, ebuf2,
        score_s2, score_b2, nullptr, 256, 256, 256, NPp, 0, 0);
    head_kernel<<<nb((long long)Bt * NPp), blk, 0, stream>>>(score_w3, score_bb3,
        ebuf2, out_score, 1, 256, NPp);
}
